// OT_Loss4_71760313581810
// MI455X (gfx1250) — matrix operand order repaired and checked by reading, not compiled, not run on hardware
//
#include <hip/hip_runtime.h>
#include <hip/hip_bf16.h>

// Sinkhorn OT loss for MI455X (gfx1250), exploiting K = Ky ⊗ Kx separability.
// One workgroup (512 thr = 16 wave32) per image; everything LDS-resident;
// matvecs recast as bf16 WMMA GEMMs with f32 accumulation.
// Round 4: GEMM1 A-operand scaling uses native V_PK_MUL_BF16 (inline asm,
// VOP3P op 42) instead of promote-to-f32 lowering.

#define SS      64
#define NBX     4096
#define NPTS    512
#define NITERS  100
#define NWAVES  16
#define BATCH   16
#define EPS16   1e-16f
#define KVSTR   20           // padded row stride (dwords) for the Kv staging buffer

typedef __attribute__((ext_vector_type(16))) __bf16 v16bf;
typedef __attribute__((ext_vector_type(8)))  __bf16 v8bf;
typedef __attribute__((ext_vector_type(8)))  float  v8f;
typedef __attribute__((ext_vector_type(4)))  unsigned int v4u;

struct __align__(64) Smem {
    __bf16 KxT[SS * NPTS];     // [gx][p]  (GEMM1 B, wd: contiguous in p)
    __bf16 KyT[SS * NPTS];     // [gy][p]  (GEMM1 A, Ky-reduce: contiguous in p)
    __bf16 Kx [NPTS * SS];     // [p][gx]  (GEMM2 A: contiguous in gx)
    __bf16 Vbf[NBX];           // v as bf16, [gy*64+gx]
    __bf16 Ubf[NPTS];          // u as bf16
    float  KvP[NPTS * KVSTR];  // staging: 16 Kv partials per point (padded rows)
    float  V[NBX];             // v (f32)
    float  Bv[NBX];            // b = normed_density flat
    float  U[NPTS];
    float  Uinit[NPTS];
    float  X[NPTS];
    float  Y[NPTS];
    float  red[NWAVES];
};   // ~281 KB of the 320 KB WGP LDS

__device__ __forceinline__ v8f zero_v8f() {
    v8f z = {0.f, 0.f, 0.f, 0.f, 0.f, 0.f, 0.f, 0.f};
    return z;
}

__device__ __forceinline__ v16bf cat8(v8bf lo, v8bf hi) {
    return __builtin_shufflevector(lo, hi, 0, 1, 2, 3, 4, 5, 6, 7,
                                           8, 9, 10, 11, 12, 13, 14, 15);
}

// 8 packed bf16 multiplies via 4x V_PK_MUL_BF16 (no clang builtin exists).
// Exact same numerics as promote/mul/truncate: bf16 products fit f32 exactly.
__device__ __forceinline__ v8bf pk_mul_bf16x8(v8bf a, v8bf b) {
    const v4u ua = __builtin_bit_cast(v4u, a);
    const v4u ub = __builtin_bit_cast(v4u, b);
    v4u uc;
    unsigned int c0, c1, c2, c3;
    asm("v_pk_mul_bf16 %0, %1, %2" : "=v"(c0) : "v"(ua[0]), "v"(ub[0]));
    asm("v_pk_mul_bf16 %0, %1, %2" : "=v"(c1) : "v"(ua[1]), "v"(ub[1]));
    asm("v_pk_mul_bf16 %0, %1, %2" : "=v"(c2) : "v"(ua[2]), "v"(ub[2]));
    asm("v_pk_mul_bf16 %0, %1, %2" : "=v"(c3) : "v"(ua[3]), "v"(ub[3]));
    uc[0] = c0; uc[1] = c1; uc[2] = c2; uc[3] = c3;
    return __builtin_bit_cast(v8bf, uc);
}

// Block-wide sum, broadcast to all threads (cold path). Deterministic order.
__device__ __forceinline__ float block_sum(float x, Smem& sm, int tid) {
    #pragma unroll
    for (int m = 16; m >= 1; m >>= 1) x += __shfl_xor(x, m, 32);
    if ((tid & 31) == 0) sm.red[tid >> 5] = x;
    __syncthreads();
    if (tid == 0) {
        float s = 0.f;
        #pragma unroll
        for (int i = 0; i < NWAVES; ++i) s += sm.red[i];
        sm.red[0] = s;
    }
    __syncthreads();
    float s = sm.red[0];
    __syncthreads();   // sm.red reused by next reduction
    return s;
}

// GEMM2 phase 1:  W = Kx @ V^T  (512x64x64, per-wave 32-row p-band), Ky-weight in
// registers, then scatter the 16 per-point partials into the padded staging buffer.
__device__ __forceinline__ void update_u_phase1(Smem& sm, int lane, int w) {
    const int pbase = 32 * w;
    const int m     = lane & 15;
    const int kb    = (lane & 16) ? 8 : 0;     // A-matrix intra-lane K pattern (ISA 7.12.2)
    const int koff  = (lane & 16) ? 16 : 0;    // unused: B loads use the same kb pattern as A

    v8f acc[2][4];
    #pragma unroll
    for (int mt = 0; mt < 2; ++mt)
        #pragma unroll
        for (int nt = 0; nt < 4; ++nt) acc[mt][nt] = zero_v8f();

    #pragma unroll
    for (int kc = 0; kc < SS; kc += 32) {         // K-dim = gx (64)
        v16bf a[2];
        #pragma unroll
        for (int mt = 0; mt < 2; ++mt) {
            const int p = pbase + 16 * mt + m;    // A row = p; row-major Kx: contiguous
            const v8bf a0 = *(const v8bf*)&sm.Kx[p * SS + kc + kb];
            const v8bf a1 = *(const v8bf*)&sm.Kx[p * SS + kc + 16 + kb];
            a[mt] = cat8(a0, a1);
        }
        #pragma unroll
        for (int nt = 0; nt < 4; ++nt) {
            // B[k=gx][n=gy] = V[gy][gx]: two 8-bf16 halves per lane at kb and 16+kb (16B aligned)
            const v16bf b = cat8(*(const v8bf*)&sm.Vbf[(16 * nt + m) * SS + kc + kb],
                                 *(const v8bf*)&sm.Vbf[(16 * nt + m) * SS + kc + 16 + kb]);
            #pragma unroll
            for (int mt = 0; mt < 2; ++mt)
                acc[mt][nt] = __builtin_amdgcn_wmma_f32_16x16x32_bf16(
                    false, a[mt], false, b, (short)0, acc[mt][nt], false, false);
        }
    }

    // Ky weighting. C layout: VGPR r -> M = r (lanes 0-15) / 8+r (lanes 16-31);
    // for fixed (mt, lane-half) the 8 r's hit consecutive p => one v8bf KyT load per (mt,nt).
    #pragma unroll
    for (int mt = 0; mt < 2; ++mt) {
        const int pstart = pbase + 16 * mt + ((lane & 16) ? 8 : 0);
        float s[8];
        #pragma unroll
        for (int r = 0; r < 8; ++r) s[r] = 0.f;
        #pragma unroll
        for (int nt = 0; nt < 4; ++nt) {
            const int gy = 16 * nt + m;
            const v8bf kyv = *(const v8bf*)&sm.KyT[gy * NPTS + pstart];
            #pragma unroll
            for (int r = 0; r < 8; ++r) s[r] += acc[mt][nt][r] * (float)kyv[r];
        }
        // Conflict-free stores: for fixed r, the 16 lanes of a half write 16
        // consecutive dwords of row (pstart+r).
        #pragma unroll
        for (int r = 0; r < 8; ++r) sm.KvP[(pstart + r) * KVSTR + m] = s[r];
    }
}

// GEMM2 phase 2 (after barrier): thread p sums its 16 staged partials
// (4 x float4, rows 16B-aligned, lane stride 20 dwords => 2-way conflicts max)
// and finishes u[p] = a / (Kv[p] + eps).
__device__ __forceinline__ void update_u_phase2(Smem& sm, int tid, float a_mass,
                                                bool store_init) {
    const float4* row = (const float4*)&sm.KvP[tid * KVSTR];
    const float4 q0 = row[0], q1 = row[1], q2 = row[2], q3 = row[3];
    const float s = ((q0.x + q0.y) + (q0.z + q0.w)) + ((q1.x + q1.y) + (q1.z + q1.w)) +
                    ((q2.x + q2.y) + (q2.z + q2.w)) + ((q3.x + q3.y) + (q3.z + q3.w));
    const float u = a_mass / (s + EPS16);
    sm.U[tid]   = u;
    sm.Ubf[tid] = (__bf16)u;
    if (store_init) sm.Uinit[tid] = u;
}

// GEMM1: R[gy,gx] = sum_p (u_p * Ky[p,gy]) * Kx[p,gx]  (64x64x512, one 16x16 tile/wave)
// mode 0: v[g] = b[g]/(R[g]+eps).   mode 1: accumulate err partial (b - R*v)^2.
__device__ __forceinline__ float gemm1(Smem& sm, int lane, int w, int mode) {
    const int ty   = w >> 2, tx = w & 3;
    const int n    = lane & 15;
    const int kb   = (lane & 16) ? 8 : 0;
    const int koff = (lane & 16) ? 16 : 0;
    const __bf16* kyRow = &sm.KyT[(16 * ty + n) * NPTS];   // A row gy (contiguous p)
    const __bf16* kxRow = &sm.KxT[(16 * tx + n) * NPTS];   // B col gx (contiguous p)

    v8f acc = zero_v8f();
    for (int kc = 0; kc < NPTS; kc += 32) {                // K-dim = p (512)
        const v8bf ky0 = *(const v8bf*)(kyRow + kc + kb);
        const v8bf ky1 = *(const v8bf*)(kyRow + kc + 16 + kb);
        const v8bf u0  = *(const v8bf*)(&sm.Ubf[kc + kb]);        // broadcast reads
        const v8bf u1  = *(const v8bf*)(&sm.Ubf[kc + 16 + kb]);
        const v16bf a  = cat8(pk_mul_bf16x8(ky0, u0),     // native packed bf16 mul
                              pk_mul_bf16x8(ky1, u1));
        const v16bf b  = cat8(*(const v8bf*)(kxRow + kc + kb),
                              *(const v8bf*)(kxRow + kc + 16 + kb));
        acc = __builtin_amdgcn_wmma_f32_16x16x32_bf16(
            false, a, false, b, (short)0, acc, false, false);
    }

    float errp = 0.f;
    #pragma unroll
    for (int r = 0; r < 8; ++r) {
        const int gy = 16 * ty + ((lane & 16) ? (8 + r) : r);
        const int g  = gy * SS + 16 * tx + n;
        const float R = acc[r];
        if (mode == 0) {
            const float v = sm.Bv[g] / (R + EPS16);
            sm.V[g]   = v;
            sm.Vbf[g] = (__bf16)v;
        } else {
            const float d = sm.Bv[g] - R * sm.V[g];
            errp += d * d;
        }
    }
    return errp;
}

__global__ void __launch_bounds__(512, 1)
sinkhorn16(const float* __restrict__ nd, const float* __restrict__ ud,
           const float* __restrict__ pts, const float* __restrict__ vpred,
           float* __restrict__ partial) {
    __shared__ Smem sm;
    const int tid  = threadIdx.x;
    const int lane = tid & 31;
    const int w    = tid >> 5;
    const int blk  = blockIdx.x;
    const float a_mass = 1.0f / (float)NPTS;

    // ---- init: b, v, and the separable kernel factors Kx (both layouts), Ky ----
    for (int g = tid; g < NBX; g += 512) {
        sm.Bv[g] = nd[blk * NBX + g];
        const float v0 = vpred[blk * NBX + g];
        sm.V[g]   = v0;
        sm.Vbf[g] = (__bf16)v0;
    }
    {
        const int p = tid;   // exactly 512 threads = 512 points
        const float xn = pts[blk * NPTS * 2 + 2 * p]     * (2.0f / 512.0f) - 1.0f;
        const float yn = pts[blk * NPTS * 2 + 2 * p + 1] * (2.0f / 512.0f) - 1.0f;
        sm.X[p] = xn;
        sm.Y[p] = yn;
        #pragma unroll 4
        for (int gc = 0; gc < SS; ++gc) {
            const float c  = (float)(gc * 8 + 4) * (2.0f / 512.0f) - 1.0f;
            const float dx = xn - c, dy = yn - c;
            const __bf16 kx = (__bf16)__expf(dx * dx * -0.1f);
            sm.KxT[gc * NPTS + p] = kx;
            sm.Kx [p * SS + gc]   = kx;
            sm.KyT[gc * NPTS + p] = (__bf16)__expf(dy * dy * -0.1f);
        }
    }
    __syncthreads();

    // ---- u_init = a / (K @ v_pred + eps) ----
    update_u_phase1(sm, lane, w);
    __syncthreads();
    update_u_phase2(sm, tid, a_mass, /*store_init=*/true);
    __syncthreads();

    // ---- 100 Sinkhorn iterations, fully LDS-resident ----
    for (int it = 0; it < NITERS; ++it) {
        gemm1(sm, lane, w, 0);                    // v = b / (u@K + eps)
        __syncthreads();
        update_u_phase1(sm, lane, w);             // u = a / (K@v + eps)
        __syncthreads();
        update_u_phase2(sm, tid, a_mass, false);
        __syncthreads();
    }

    // ---- err = sum((b - u@K * v)^2) with the FINAL u (extra GEMM1 pass) ----
    float errp = gemm1(sm, lane, w, 1);
    __syncthreads();

    // ---- scalar losses ----
    float sumv_p = 0.f, cnt_p = 0.f, S1_p = 0.f, ot_p = 0.f;
    for (int g = tid; g < NBX; g += 512) {
        const float v    = sm.V[g];
        const float beta = 10.0f * __logf(v + EPS16);
        const float udg  = ud[blk * NBX + g];
        sumv_p += v;
        cnt_p  += udg;
        S1_p   += udg * beta;
        ot_p   += sm.Bv[g] * beta;
    }
    const float sumv  = block_sum(sumv_p, sm, tid);
    const float cnt   = block_sum(cnt_p,  sm, tid);
    const float S1    = block_sum(S1_p,   sm, tid);
    const float ot    = block_sum(ot_p,   sm, tid);
    const float denom = cnt * cnt + 1e-8f;

    float lossv_p = 0.f, loss_p = 0.f;
    for (int g = tid; g < NBX; g += 512) {
        const float v    = sm.V[g];
        const float beta = 10.0f * __logf(v + EPS16);
        const float vn   = v / (sumv + EPS16);
        lossv_p += vn * (__logf(vn) - __logf(vpred[blk * NBX + g]));
        const float udg = ud[blk * NBX + g];
        loss_p += udg * ((cnt / denom) * beta - S1 / denom);
    }
    const float lossv = block_sum(lossv_p, sm, tid);
    const float loss  = block_sum(loss_p,  sm, tid);

    // lossu = -(u_init . u) / (||u_init|| * ||u||)
    const float dotu = block_sum(sm.Uinit[tid] * sm.U[tid],     sm, tid);
    const float nn1  = block_sum(sm.Uinit[tid] * sm.Uinit[tid], sm, tid);
    const float nn2  = block_sum(sm.U[tid] * sm.U[tid],         sm, tid);
    const float lossu =
        -(dotu / (fmaxf(sqrtf(nn1), 1e-8f) * fmaxf(sqrtf(nn2), 1e-8f)));

    // wd = sum_{p,g} dis * u_p * K * v_g   (one p per thread; dis recomputed exactly)
    float wd_p = 0.f;
    {
        const int p = tid;
        const float up = sm.U[p], xn = sm.X[p], yn = sm.Y[p];
        for (int gy = 0; gy < SS; ++gy) {
            const float cy  = (float)(gy * 8 + 4) * (2.0f / 512.0f) - 1.0f;
            const float dy2 = (yn - cy) * (yn - cy);
            const float ky  = (float)sm.KyT[gy * NPTS + p];
            float inner = 0.f;
            #pragma unroll 8
            for (int gx = 0; gx < SS; ++gx) {
                const float cx  = (float)(gx * 8 + 4) * (2.0f / 512.0f) - 1.0f;
                const float dx2 = (xn - cx) * (xn - cx);
                inner += (dy2 + dx2) * (float)sm.KxT[gx * NPTS + p] * sm.V[gy * SS + gx];
            }
            wd_p += ky * inner;
        }
        wd_p *= up;
    }
    const float wd  = block_sum(wd_p, sm, tid);
    const float err = block_sum(errp, sm, tid);

    if (tid == 0) {
        float* o = partial + blk * 6;
        o[0] = loss; o[1] = lossv; o[2] = lossu; o[3] = wd; o[4] = ot; o[5] = err;
    }
}

// Deterministic final reduction over the 16 per-image partials (fixed order).
__global__ void reduce6(const float* __restrict__ partial, float* __restrict__ out) {
    const int i = threadIdx.x;
    if (i < 6) {
        float s = 0.f;
        for (int b = 0; b < BATCH; ++b) s += partial[b * 6 + i];
        out[i] = s;
    }
}

extern "C" void kernel_launch(void* const* d_in, const int* in_sizes, int n_in,
                              void* d_out, int out_size, void* d_ws, size_t ws_size,
                              hipStream_t stream) {
    (void)in_sizes; (void)n_in; (void)out_size; (void)ws_size;
    const float* nd    = (const float*)d_in[0];
    const float* ud    = (const float*)d_in[1];
    const float* pts   = (const float*)d_in[2];
    const float* vpred = (const float*)d_in[3];
    float* partial = (float*)d_ws;   // 16 * 6 floats

    sinkhorn16<<<BATCH, 512, 0, stream>>>(nd, ud, pts, vpred, partial);
    reduce6<<<1, 32, 0, stream>>>(partial, (float*)d_out);
}
